// SpatiallyAwareISAAQ_15075335209470
// MI455X (gfx1250) — matrix operand order repaired and checked by reading, not compiled, not run on hardware
//
#include <hip/hip_runtime.h>
#include <math.h>

// ---------------------------------------------------------------------------
// SpatiallyAwareISAAQ forward for MI455X (gfx1250, wave32, WMMA).
// All heavy contractions run through v_wmma_f32_16x16x32_bf16 (bf16 in, f32
// accumulate).  Attention is flash-style (online softmax) so the 64x16x244x244
// score tensor is never materialized.  FFN GELU and QKV bias+pack are fused
// into the GEMM epilogue.  GEMM A-tile staging uses the CDNA5 async
// global->LDS copy (GLOBAL_LOAD_ASYNC_TO_LDS_B128, ASYNCcnt) overlapped with
// the manual B-tile transpose, plus global_prefetch for the next B tile.
//
// Input pointer map (setup_inputs() dict insertion order, nested dicts /
// layer list flattened in insertion order):
//   0 sequence_output [16,4,180,1024] f32     1 vis_feats [16,64,2048] f32
//   2 coords [16,64,4] f32                    3 ocr_text_feats [16,64,1024] f32
//   4 vis_adj [16,64,64] i32                  5 text_type_ids [16,4,180] i32
//   6 type_emb [3,1024]   7 vis_W [3076,1024] 8 vis_b  9 vis_g  10 vis_beta
//  11 txt_W [1024,1024]  12 txt_b 13 txt_g 14 txt_beta 15 rel_bias [15,16]
//  16..31 layer0: Wq,bq,Wk,bk,Wv,bv,Wo,bo,g1,b1,W1,bf1,W2,bf2,g2,b2
//  32..47 layer1 (same order)
//  48 att1_W [2048,1024] 49 att1_b 50 gate_W 51 gate_b 52 aw_W 53 aw_b
//  54 cls_W [1024,1]     55 cls_b
// ---------------------------------------------------------------------------

typedef __attribute__((ext_vector_type(16))) __bf16 v16bf;
typedef __attribute__((ext_vector_type(8)))  float  v8f;

union Frag16 {
  v16bf v;
  unsigned int u[8];
  unsigned short s[16];
};

__device__ __forceinline__ unsigned short f2bf(float f) {
  unsigned int u = __float_as_uint(f);
  u += 0x7FFFu + ((u >> 16) & 1u);   // round-to-nearest-even
  return (unsigned short)(u >> 16);
}
__device__ __forceinline__ float bf2f(unsigned short h) {
  return __uint_as_float(((unsigned int)h) << 16);
}
__device__ __forceinline__ v8f vzero8() {
  v8f z = {0.f, 0.f, 0.f, 0.f, 0.f, 0.f, 0.f, 0.f};
  return z;
}

// Problem constants
#define N_TOK 244        // 64 vis + 180 txt tokens
#define N_PAD 256        // padded key/query count for WMMA tiles
#define BC    64         // B*C
#define NHEAD 16
#define DHEAD 64
#define DMODEL 1024
#define M_TOK (BC * N_TOK)   // 15616 token rows (multiple of 64)

// ---------------------------------------------------------------------------
// Generic bf16 WMMA GEMM: C[M,N] = A[M,K] * B[K,N]
// Block tile 64x128, 8 waves (4 M-strips x 2 N-strips), per-wave 16x64.
// ---------------------------------------------------------------------------
#define TILE_M 64
#define TILE_N 128
#define LDA 40   // padded LDS strides (80B rows -> 16B aligned, low conflicts)
#define LDB 40

enum { EPI_F32 = 0, EPI_GELU_BF16 = 1, EPI_QK_PACK = 2, EPI_VT_PACK = 3 };

__global__ __launch_bounds__(256) void gemm_bf16_kernel(
    const unsigned short* __restrict__ A,
    const unsigned short* __restrict__ B,
    int M, int N, int K, int epi,
    float* __restrict__ Cf,
    unsigned short* __restrict__ Cb,
    const float* __restrict__ bias)
{
  __shared__ unsigned short Asm[TILE_M * LDA];
  __shared__ unsigned short Bsm[TILE_N * LDB];

  const int bm = blockIdx.y * TILE_M;
  const int bn = blockIdx.x * TILE_N;
  const int tid  = threadIdx.x;
  const int wave = tid >> 5;
  const int lane = tid & 31;
  const int wm = wave & 3;    // 16-row strip
  const int wn = wave >> 2;   // 64-col strip
  const int half = lane >> 4;
  const int l16  = lane & 15;

  // Per-thread fixed staging slots (computed once).
  const int ea = tid * 8;
  const int ar = ea >> 5, ac = ea & 31;
  const unsigned lds_a = (unsigned)(size_t)&Asm[ar * LDA + ac];

  v8f acc[4];
#pragma unroll
  for (int j = 0; j < 4; ++j) acc[j] = vzero8();

  for (int k0 = 0; k0 < K; k0 += 32) {
    // ---- Stage A tile (64x32 bf16): CDNA5 async global->LDS, 16B per lane.
    {
      const unsigned short* src = A + (size_t)(bm + ar) * K + (k0 + ac);
      asm volatile("global_load_async_to_lds_b128 %0, %1, off"
                   :: "v"(lds_a), "v"((unsigned long long)(size_t)src)
                   : "memory");
    }
    // ---- Stage B tile transposed (manual, overlaps with async A copy):
    //      B[k0..+31][bn..+127] -> Bsm[n][k]
#pragma unroll
    for (int p = 0; p < 2; ++p) {
      int e = tid * 8 + p * 2048;
      int kk = e >> 7, n = e & 127;
      const unsigned short* src = B + (size_t)(k0 + kk) * N + (bn + n);
      uint4 d = *(const uint4*)src;
      const unsigned short* ds = (const unsigned short*)&d;
#pragma unroll
      for (int ii = 0; ii < 8; ++ii)
        Bsm[(n + ii) * LDB + kk] = ds[ii];
      if (k0 + 32 < K)   // prefetch next K-step of B (streamed, no reuse)
        __builtin_prefetch(src + (size_t)32 * N, 0, 1);
    }
    // Wait for this wave's async LDS writes, then workgroup barrier.
    asm volatile("s_wait_asynccnt 0" ::: "memory");
    __syncthreads();

    // A fragment (16x32): lane<16 -> cols 0..15; lane>=16 -> cols 16..31
    Frag16 a;
    {
      const unsigned short* base = &Asm[(16 * wm + l16) * LDA];
      const int c1 = half ? 16 : 0;
      const int c2 = half ? 24 : 8;
#pragma unroll
      for (int j = 0; j < 4; ++j) {
        a.u[j]     = *(const unsigned int*)(base + c1 + 2 * j);
        a.u[4 + j] = *(const unsigned int*)(base + c2 + 2 * j);
      }
    }
#pragma unroll
    for (int j = 0; j < 4; ++j) {
      // B fragment (32x16): col n = l16; lane<16 rows 0..15, lane>=16 rows 16..31
      Frag16 b;
      const unsigned short* base = &Bsm[(64 * wn + 16 * j + l16) * LDB + half * 16];
#pragma unroll
      for (int i = 0; i < 8; ++i)
        b.u[i] = *(const unsigned int*)(base + 2 * i);
      acc[j] = __builtin_amdgcn_wmma_f32_16x16x32_bf16(
          false, a.v, false, b.v, (short)0, acc[j], false, false);
    }
    __syncthreads();
  }

  // Epilogue: element (m,n) -> row = r + 8*half, col = l16 within 16x16 tile
#pragma unroll
  for (int j = 0; j < 4; ++j) {
#pragma unroll
    for (int r = 0; r < 8; ++r) {
      int row = bm + 16 * wm + r + 8 * half;
      int col = bn + 64 * wn + 16 * j + l16;
      float v = acc[j][r];
      if (epi == EPI_F32) {
        Cf[(size_t)row * N + col] = v;
      } else if (epi == EPI_GELU_BF16) {
        v += bias[col];
        float t = tanhf(0.7978845608028654f * (v + 0.044715f * v * v * v));
        Cb[(size_t)row * N + col] = f2bf(0.5f * v * (1.f + t));
      } else {
        v += bias[col];
        int bc = row / N_TOK;
        int n  = row - bc * N_TOK;
        int h  = col >> 6;
        int dh = col & 63;
        if (epi == EPI_QK_PACK)
          Cb[(((size_t)bc * NHEAD + h) * N_PAD + n) * DHEAD + dh] = f2bf(v);
        else
          Cb[(((size_t)bc * NHEAD + h) * DHEAD + dh) * N_PAD + n] = f2bf(v);
      }
    }
  }
}

// ---------------------------------------------------------------------------
// Flash attention: one wave per (q-tile of 16, head, bc).  Keys padded to 256
// with -1e30 bias.  Scores and P*V via WMMA bf16.
// ---------------------------------------------------------------------------
__device__ __forceinline__ float attn_bias_val(
    int q, int k, int b, const float* relbS, const int* __restrict__ vis_adj)
{
  if (k >= N_TOK) return -1e30f;
  int a;
  if (q < 64 && k < 64)      a = vis_adj[((size_t)b * 64 + q) * 64 + k];
  else if (q < 64 || k < 64) a = 13;
  else                       a = 14;
  return relbS[a];
}

__global__ __launch_bounds__(32) void attn_kernel(
    const unsigned short* __restrict__ qb,
    const unsigned short* __restrict__ kb,
    const unsigned short* __restrict__ vt,
    const float* __restrict__ rel_bias,
    const int* __restrict__ vis_adj,
    unsigned short* __restrict__ ctx)
{
  const int qt = blockIdx.x;
  const int h  = blockIdx.y;
  const int bc = blockIdx.z;
  const int b  = bc >> 2;
  const int lane = threadIdx.x;
  const int half = lane >> 4;
  const int l16  = lane & 15;
  const int q0 = qt * 16;

  __shared__ float relbS[16];
  __shared__ unsigned short pt[16 * 34];   // P tile, stride 34 (68B rows)
  if (lane < 15) relbS[lane] = rel_bias[lane * NHEAD + h];
  if (lane == 15) relbS[15] = 0.f;
  __syncthreads();

  const unsigned short* Q  = qb + (((size_t)bc * NHEAD + h) * N_PAD) * DHEAD;
  const unsigned short* Kp = kb + (((size_t)bc * NHEAD + h) * N_PAD) * DHEAD;
  const unsigned short* Vt = vt + (((size_t)bc * NHEAD + h) * DHEAD) * N_PAD;

  // Q fragments (16 rows x 64 dh as two 16x32 A frags)
  Frag16 qa[2];
  {
    const unsigned short* qr = Q + (size_t)(q0 + l16) * DHEAD;
#pragma unroll
    for (int hk = 0; hk < 2; ++hk) {
      const int c1 = 32 * hk + (half ? 16 : 0);
      const int c2 = 32 * hk + (half ? 24 : 8);
#pragma unroll
      for (int j = 0; j < 4; ++j) {
        qa[hk].u[j]     = *(const unsigned int*)(qr + c1 + 2 * j);
        qa[hk].u[4 + j] = *(const unsigned int*)(qr + c2 + 2 * j);
      }
    }
  }

  v8f o[4];
#pragma unroll
  for (int j = 0; j < 4; ++j) o[j] = vzero8();
  float mi[8], li[8];
#pragma unroll
  for (int r = 0; r < 8; ++r) { mi[r] = -1e30f; li[r] = 0.f; }

  for (int kt = 0; kt < 8; ++kt) {
    const int k0 = kt * 32;
    v8f s0 = vzero8(), s1 = vzero8();
#pragma unroll
    for (int hk = 0; hk < 2; ++hk) {
      const int dhb = 32 * hk + half * 16;
      Frag16 bf;
      const unsigned short* kr = Kp + (size_t)(k0 + l16) * DHEAD + dhb;
#pragma unroll
      for (int i = 0; i < 8; ++i) bf.u[i] = *(const unsigned int*)(kr + 2 * i);
      s0 = __builtin_amdgcn_wmma_f32_16x16x32_bf16(
          false, qa[hk].v, false, bf.v, (short)0, s0, false, false);
      const unsigned short* kr2 = Kp + (size_t)(k0 + 16 + l16) * DHEAD + dhb;
#pragma unroll
      for (int i = 0; i < 8; ++i) bf.u[i] = *(const unsigned int*)(kr2 + 2 * i);
      s1 = __builtin_amdgcn_wmma_f32_16x16x32_bf16(
          false, qa[hk].v, false, bf.v, (short)0, s1, false, false);
    }

    const int kA = k0 + l16;
    const int kB = k0 + 16 + l16;
    float p0[8], p1[8], corr[8];
#pragma unroll
    for (int r = 0; r < 8; ++r) {
      const int q = q0 + r + 8 * half;
      float v0 = s0[r] * 0.125f + attn_bias_val(q, kA, b, relbS, vis_adj);
      float v1 = s1[r] * 0.125f + attn_bias_val(q, kB, b, relbS, vis_adj);
      float rm = fmaxf(v0, v1);
#pragma unroll
      for (int off = 1; off < 16; off <<= 1)
        rm = fmaxf(rm, __shfl_xor(rm, off, 32));
      float mn = fmaxf(mi[r], rm);
      corr[r] = __expf(mi[r] - mn);
      mi[r] = mn;
      p0[r] = __expf(v0 - mn);
      p1[r] = __expf(v1 - mn);
      float rs = p0[r] + p1[r];
#pragma unroll
      for (int off = 1; off < 16; off <<= 1)
        rs += __shfl_xor(rs, off, 32);
      li[r] = li[r] * corr[r] + rs;
    }
#pragma unroll
    for (int j = 0; j < 4; ++j)
#pragma unroll
      for (int r = 0; r < 8; ++r)
        o[j][r] *= corr[r];

    // Transpose P through LDS into A-fragment layout.
    __syncthreads();
#pragma unroll
    for (int r = 0; r < 8; ++r) {
      pt[(r + 8 * half) * 34 + l16]      = f2bf(p0[r]);
      pt[(r + 8 * half) * 34 + 16 + l16] = f2bf(p1[r]);
    }
    __syncthreads();
    Frag16 pa;
    {
      const unsigned short* base = &pt[l16 * 34];
      const int c1 = half ? 16 : 0;
      const int c2 = half ? 24 : 8;
#pragma unroll
      for (int j = 0; j < 4; ++j) {
        pa.u[j]     = *(const unsigned int*)(base + c1 + 2 * j);
        pa.u[4 + j] = *(const unsigned int*)(base + c2 + 2 * j);
      }
    }
#pragma unroll
    for (int j = 0; j < 4; ++j) {
      Frag16 vb;
      const unsigned short* vr = Vt + (size_t)(16 * j + l16) * N_PAD + k0 + half * 16;
#pragma unroll
      for (int i = 0; i < 8; ++i) vb.u[i] = *(const unsigned int*)(vr + 2 * i);
      o[j] = __builtin_amdgcn_wmma_f32_16x16x32_bf16(
          false, pa.v, false, vb.v, (short)0, o[j], false, false);
    }
  }

#pragma unroll
  for (int r = 0; r < 8; ++r) {
    const int q = q0 + r + 8 * half;
    if (q < N_TOK) {
      const float inv = 1.f / li[r];
      const size_t base = ((size_t)bc * N_TOK + q) * DMODEL + h * DHEAD;
#pragma unroll
      for (int j = 0; j < 4; ++j)
        ctx[base + 16 * j + l16] = f2bf(o[j][r] * inv);
    }
  }
}

// ---------------------------------------------------------------------------
// Row-wise LayerNorm epilogues.  mode 0: vis (relu(LN)+type_emb[0]);
// mode 1: txt (LN + type_emb[types], in place); mode 2: residual LN.
// ---------------------------------------------------------------------------
__global__ __launch_bounds__(256) void ln_rows_kernel(
    const float* __restrict__ Cin, const float* __restrict__ bias,
    const float* __restrict__ resid,
    const float* __restrict__ gamma, const float* __restrict__ beta,
    const float* __restrict__ type_emb, const int* __restrict__ ttids,
    int mode, float* __restrict__ outF, unsigned short* __restrict__ outB)
{
  const int row = blockIdx.x;
  const int t = threadIdx.x;
  float v[4];
  float s = 0.f, s2 = 0.f;
#pragma unroll
  for (int i = 0; i < 4; ++i) {
    int d = t + 256 * i;
    float x = Cin[(size_t)row * DMODEL + d] + bias[d];
    if (mode == 2) x += resid[(size_t)row * DMODEL + d];
    v[i] = x; s += x; s2 += x * x;
  }
  __shared__ float r1[256], r2[256];
  r1[t] = s; r2[t] = s2;
  __syncthreads();
  for (int o = 128; o > 0; o >>= 1) {
    if (t < o) { r1[t] += r1[t + o]; r2[t] += r2[t + o]; }
    __syncthreads();
  }
  const float mean = r1[0] * (1.f / 1024.f);
  const float var  = r2[0] * (1.f / 1024.f) - mean * mean;
  const float rstd = rsqrtf(var + 1e-5f);
  int te = -1;
  if (mode == 0) te = 0;
  else if (mode == 1) {
    int si = row % 180;
    te = (si == 0) ? 1 : (ttids[row] + 1);
  }
#pragma unroll
  for (int i = 0; i < 4; ++i) {
    int d = t + 256 * i;
    float y = (v[i] - mean) * rstd * gamma[d] + beta[d];
    if (mode == 0) y = fmaxf(y, 0.f);
    if (te >= 0) y += type_emb[te * DMODEL + d];
    outF[(size_t)row * DMODEL + d] = y;
    if (outB) outB[(size_t)row * DMODEL + d] = f2bf(y);
  }
}

// ---------------------------------------------------------------------------
// Small utility / assembly kernels
// ---------------------------------------------------------------------------
__global__ void cvt_bf16_kernel(const float* __restrict__ in,
                                unsigned short* __restrict__ out, size_t n)
{
  size_t i = (size_t)blockIdx.x * blockDim.x + threadIdx.x;
  const size_t st = (size_t)gridDim.x * blockDim.x;
  for (; i < n; i += st) out[i] = f2bf(in[i]);
}

__global__ void cvt_pad_rows_kernel(const float* __restrict__ in,
                                    unsigned short* __restrict__ out,
                                    int rows_in, int cols, size_t n)
{
  size_t i = (size_t)blockIdx.x * blockDim.x + threadIdx.x;
  const size_t st = (size_t)gridDim.x * blockDim.x;
  for (; i < n; i += st) {
    int r = (int)(i / cols);
    out[i] = (r < rows_in) ? f2bf(in[i]) : (unsigned short)0;
  }
}

__global__ void build_avis_kernel(const float* __restrict__ vf,
                                  const float* __restrict__ co,
                                  const float* __restrict__ ocr,
                                  unsigned short* __restrict__ out)
{
  const size_t n = (size_t)1024 * 3104;
  size_t i = (size_t)blockIdx.x * blockDim.x + threadIdx.x;
  const size_t st = (size_t)gridDim.x * blockDim.x;
  for (; i < n; i += st) {
    int row = (int)(i / 3104);
    int c = (int)(i % 3104);
    float v;
    if (c < 2048)       v = vf[(size_t)row * 2048 + c];
    else if (c < 2052)  v = co[(size_t)row * 4 + (c - 2048)];
    else if (c < 3076)  v = ocr[(size_t)row * 1024 + (c - 2052)];
    else                v = 0.f;
    out[i] = f2bf(v);
  }
}

__global__ void assemble_x_kernel(const float* __restrict__ vis,
                                  const float* __restrict__ txt,
                                  float* __restrict__ xF,
                                  unsigned short* __restrict__ xB)
{
  const size_t n = (size_t)BC * N_TOK * DMODEL;
  size_t i = (size_t)blockIdx.x * blockDim.x + threadIdx.x;
  const size_t st = (size_t)gridDim.x * blockDim.x;
  for (; i < n; i += st) {
    int bc = (int)(i / ((size_t)N_TOK * DMODEL));
    size_t rem = i - (size_t)bc * N_TOK * DMODEL;
    int tok = (int)(rem / DMODEL);
    int d = (int)(rem % DMODEL);
    float v = (tok < 64)
        ? vis[(((size_t)(bc >> 2)) * 64 + tok) * DMODEL + d]
        : txt[((size_t)bc * 180 + (tok - 64)) * DMODEL + d];
    xF[i] = v;
    xB[i] = f2bf(v);
  }
}

__global__ void build_attin_kernel(const float* __restrict__ xF,
                                   unsigned short* __restrict__ out)
{
  const size_t n = (size_t)BC * 64 * 2048;
  size_t i = (size_t)blockIdx.x * blockDim.x + threadIdx.x;
  const size_t st = (size_t)gridDim.x * blockDim.x;
  for (; i < n; i += st) {
    int bc = (int)(i / ((size_t)64 * 2048));
    size_t rem = i - (size_t)bc * 64 * 2048;
    int nv = (int)(rem / 2048);
    int d = (int)(rem % 2048);
    float v = (d < 1024)
        ? xF[((size_t)bc * N_TOK + nv) * DMODEL + d]
        : xF[((size_t)bc * N_TOK + 64) * DMODEL + (d - 1024)];
    out[i] = f2bf(v);
  }
}

__global__ __launch_bounds__(256) void att_finish_kernel(
    const float* __restrict__ c, const float* __restrict__ b1,
    const float* __restrict__ gw, const float* __restrict__ gb,
    const float* __restrict__ aw, const float* __restrict__ ab,
    float* __restrict__ sout)
{
  const int row = blockIdx.x;
  const int t = threadIdx.x;
  float dg = 0.f, da = 0.f;
#pragma unroll
  for (int i = 0; i < 4; ++i) {
    int d = t + 256 * i;
    float x = tanhf(c[(size_t)row * 1024 + d] + b1[d]);
    dg += x * gw[d];
    da += x * aw[d];
  }
  __shared__ float r1[256], r2[256];
  r1[t] = dg; r2[t] = da;
  __syncthreads();
  for (int o = 128; o > 0; o >>= 1) {
    if (t < o) { r1[t] += r1[t + o]; r2[t] += r2[t + o]; }
    __syncthreads();
  }
  if (t == 0) {
    float gate = 1.f / (1.f + __expf(-(r1[0] + gb[0])));
    sout[row] = gate * r2[0] + ab[0];
  }
}

__global__ __launch_bounds__(256) void pool_kernel(
    const float* __restrict__ xF, const float* __restrict__ sarr,
    const float* __restrict__ clsW, const float* __restrict__ clsb,
    float* __restrict__ out)
{
  const int bc = blockIdx.x;
  const int t = threadIdx.x;
  __shared__ float w[64];
  __shared__ float red[256];
  if (t < 64) w[t] = sarr[bc * 64 + t];
  __syncthreads();
  if (t == 0) {
    float m = -3.4e38f;
    for (int i = 0; i < 64; ++i) m = fmaxf(m, w[i]);
    float sm = 0.f;
    for (int i = 0; i < 64; ++i) { w[i] = __expf(w[i] - m); sm += w[i]; }
    float inv = 1.f / sm;
    for (int i = 0; i < 64; ++i) w[i] *= inv;
  }
  __syncthreads();
  float part = 0.f;
  for (int d = t; d < 1024; d += 256) {
    float vp = 0.f;
    for (int nv = 0; nv < 64; ++nv)
      vp += w[nv] * xF[((size_t)bc * N_TOK + nv) * DMODEL + d];
    part += xF[((size_t)bc * N_TOK + 64) * DMODEL + d] * vp * clsW[d];
  }
  red[t] = part;
  __syncthreads();
  for (int o = 128; o > 0; o >>= 1) {
    if (t < o) red[t] += red[t + o];
    __syncthreads();
  }
  if (t == 0) out[bc] = red[0] + clsb[0];
}

// ---------------------------------------------------------------------------
// Host orchestration
// ---------------------------------------------------------------------------
static unsigned gs_blocks(size_t n) {
  size_t b = (n + 255) / 256;
  if (b > 16384) b = 16384;
  return (unsigned)b;
}

extern "C" void kernel_launch(void* const* d_in, const int* in_sizes, int n_in,
                              void* d_out, int out_size, void* d_ws, size_t ws_size,
                              hipStream_t stream)
{
  (void)in_sizes; (void)n_in; (void)out_size; (void)ws_size;

  const float* seq      = (const float*)d_in[0];
  const float* vis_feat = (const float*)d_in[1];
  const float* coords   = (const float*)d_in[2];
  const float* ocr      = (const float*)d_in[3];
  const int*   vis_adj  = (const int*)d_in[4];
  const int*   ttids    = (const int*)d_in[5];
  const float* type_emb = (const float*)d_in[6];
  const float* vis_W    = (const float*)d_in[7];
  const float* vis_b    = (const float*)d_in[8];
  const float* vis_g    = (const float*)d_in[9];
  const float* vis_beta = (const float*)d_in[10];
  const float* txt_W    = (const float*)d_in[11];
  const float* txt_b    = (const float*)d_in[12];
  const float* txt_g    = (const float*)d_in[13];
  const float* txt_beta = (const float*)d_in[14];
  const float* rel_bias = (const float*)d_in[15];
  struct Layer {
    const float *Wq,*bq,*Wk,*bk,*Wv,*bv,*Wo,*bo,*g1,*b1,*W1,*bf1,*W2,*bf2,*g2,*b2;
  } L[2];
  for (int l = 0; l < 2; ++l) {
    int base = 16 + 16 * l;
    L[l].Wq  = (const float*)d_in[base + 0];  L[l].bq  = (const float*)d_in[base + 1];
    L[l].Wk  = (const float*)d_in[base + 2];  L[l].bk  = (const float*)d_in[base + 3];
    L[l].Wv  = (const float*)d_in[base + 4];  L[l].bv  = (const float*)d_in[base + 5];
    L[l].Wo  = (const float*)d_in[base + 6];  L[l].bo  = (const float*)d_in[base + 7];
    L[l].g1  = (const float*)d_in[base + 8];  L[l].b1  = (const float*)d_in[base + 9];
    L[l].W1  = (const float*)d_in[base + 10]; L[l].bf1 = (const float*)d_in[base + 11];
    L[l].W2  = (const float*)d_in[base + 12]; L[l].bf2 = (const float*)d_in[base + 13];
    L[l].g2  = (const float*)d_in[base + 14]; L[l].b2  = (const float*)d_in[base + 15];
  }
  const float* att1_W = (const float*)d_in[48];
  const float* att1_b = (const float*)d_in[49];
  const float* gate_W = (const float*)d_in[50];
  const float* gate_b = (const float*)d_in[51];
  const float* aw_W   = (const float*)d_in[52];
  const float* aw_b   = (const float*)d_in[53];
  const float* cls_W  = (const float*)d_in[54];
  const float* cls_b  = (const float*)d_in[55];

  // Workspace bump allocator (deterministic layout every call).
  char* ws = (char*)d_ws;
  size_t off = 0;
  auto alloc = [&](size_t bytes) -> char* {
    char* p = ws + off;
    off = (off + bytes + 255) & ~(size_t)255;
    return p;
  };
  typedef unsigned short u16;

  u16* wb_vis  = (u16*)alloc((size_t)3104 * 1024 * 2);
  u16* wb_txt  = (u16*)alloc((size_t)1024 * 1024 * 2);
  u16* wb_q[2], *wb_k[2], *wb_v[2], *wb_o[2], *wb_1[2], *wb_2[2];
  for (int l = 0; l < 2; ++l) {
    wb_q[l] = (u16*)alloc((size_t)1024 * 1024 * 2);
    wb_k[l] = (u16*)alloc((size_t)1024 * 1024 * 2);
    wb_v[l] = (u16*)alloc((size_t)1024 * 1024 * 2);
    wb_o[l] = (u16*)alloc((size_t)1024 * 1024 * 2);
    wb_1[l] = (u16*)alloc((size_t)1024 * 4096 * 2);
    wb_2[l] = (u16*)alloc((size_t)4096 * 1024 * 2);
  }
  u16*   wb_att = (u16*)alloc((size_t)2048 * 1024 * 2);
  u16*   a_vis  = (u16*)alloc((size_t)1024 * 3104 * 2);
  u16*   seq_bf = (u16*)alloc((size_t)11520 * 1024 * 2);
  float* cbuf   = (float*)alloc((size_t)M_TOK * 1024 * 4);
  float* visf   = (float*)alloc((size_t)1024 * 1024 * 4);
  float* xf     = (float*)alloc((size_t)M_TOK * 1024 * 4);
  u16*   xb     = (u16*)alloc((size_t)M_TOK * 1024 * 2);
  u16*   qb     = (u16*)alloc((size_t)BC * NHEAD * N_PAD * DHEAD * 2);
  u16*   kb     = (u16*)alloc((size_t)BC * NHEAD * N_PAD * DHEAD * 2);
  u16*   vtb    = (u16*)alloc((size_t)BC * NHEAD * DHEAD * N_PAD * 2);
  u16*   ctx    = (u16*)alloc((size_t)M_TOK * 1024 * 2);
  u16*   hbuf   = (u16*)alloc((size_t)M_TOK * 4096 * 2);
  u16*   attin  = (u16*)alloc((size_t)4096 * 2048 * 2);
  float* sbuf   = (float*)alloc((size_t)4096 * 4);

  const int TPB = 256;
  auto gemm = [&](const u16* A, const u16* B, int M, int N, int K,
                  int epi, float* Cf, u16* Cb, const float* bias) {
    gemm_bf16_kernel<<<dim3(N / TILE_N, M / TILE_M), TPB, 0, stream>>>(
        A, B, M, N, K, epi, Cf, Cb, bias);
  };

  // --- weight conversions (f32 -> bf16) ---
  cvt_pad_rows_kernel<<<gs_blocks((size_t)3104 * 1024), TPB, 0, stream>>>(
      vis_W, wb_vis, 3076, 1024, (size_t)3104 * 1024);
  cvt_bf16_kernel<<<gs_blocks((size_t)1024 * 1024), TPB, 0, stream>>>(
      txt_W, wb_txt, (size_t)1024 * 1024);
  for (int l = 0; l < 2; ++l) {
    cvt_bf16_kernel<<<gs_blocks((size_t)1024 * 1024), TPB, 0, stream>>>(L[l].Wq, wb_q[l], (size_t)1024 * 1024);
    cvt_bf16_kernel<<<gs_blocks((size_t)1024 * 1024), TPB, 0, stream>>>(L[l].Wk, wb_k[l], (size_t)1024 * 1024);
    cvt_bf16_kernel<<<gs_blocks((size_t)1024 * 1024), TPB, 0, stream>>>(L[l].Wv, wb_v[l], (size_t)1024 * 1024);
    cvt_bf16_kernel<<<gs_blocks((size_t)1024 * 1024), TPB, 0, stream>>>(L[l].Wo, wb_o[l], (size_t)1024 * 1024);
    cvt_bf16_kernel<<<gs_blocks((size_t)1024 * 4096), TPB, 0, stream>>>(L[l].W1, wb_1[l], (size_t)1024 * 4096);
    cvt_bf16_kernel<<<gs_blocks((size_t)4096 * 1024), TPB, 0, stream>>>(L[l].W2, wb_2[l], (size_t)4096 * 1024);
  }
  cvt_bf16_kernel<<<gs_blocks((size_t)2048 * 1024), TPB, 0, stream>>>(
      att1_W, wb_att, (size_t)2048 * 1024);
  cvt_bf16_kernel<<<gs_blocks((size_t)11520 * 1024), TPB, 0, stream>>>(
      seq, seq_bf, (size_t)11520 * 1024);
  build_avis_kernel<<<gs_blocks((size_t)1024 * 3104), TPB, 0, stream>>>(
      vis_feat, coords, ocr, a_vis);

  // --- vis embedding: relu(LN(vis_in @ vis_W + b)) + type_emb[0] ---
  gemm(a_vis, wb_vis, 1024, 1024, 3104, EPI_F32, cbuf, nullptr, nullptr);
  ln_rows_kernel<<<1024, TPB, 0, stream>>>(
      cbuf, vis_b, nullptr, vis_g, vis_beta, type_emb, nullptr, 0, visf, nullptr);

  // --- txt projection: LN(seq @ txt_W + b) + type_emb[types] (in place) ---
  gemm(seq_bf, wb_txt, 11520, 1024, 1024, EPI_F32, cbuf, nullptr, nullptr);
  ln_rows_kernel<<<11520, TPB, 0, stream>>>(
      cbuf, txt_b, nullptr, txt_g, txt_beta, type_emb, ttids, 1, cbuf, nullptr);

  assemble_x_kernel<<<gs_blocks((size_t)BC * N_TOK * DMODEL), TPB, 0, stream>>>(
      visf, cbuf, xf, xb);

  // --- transformer layers ---
  for (int l = 0; l < 2; ++l) {
    gemm(xb, wb_q[l], M_TOK, 1024, 1024, EPI_QK_PACK, nullptr, qb, L[l].bq);
    gemm(xb, wb_k[l], M_TOK, 1024, 1024, EPI_QK_PACK, nullptr, kb, L[l].bk);
    gemm(xb, wb_v[l], M_TOK, 1024, 1024, EPI_VT_PACK, nullptr, vtb, L[l].bv);
    attn_kernel<<<dim3(16, NHEAD, BC), 32, 0, stream>>>(
        qb, kb, vtb, rel_bias, vis_adj, ctx);
    gemm(ctx, wb_o[l], M_TOK, 1024, 1024, EPI_F32, cbuf, nullptr, nullptr);
    ln_rows_kernel<<<M_TOK, TPB, 0, stream>>>(
        cbuf, L[l].bo, xf, L[l].g1, L[l].b1, nullptr, nullptr, 2, xf, xb);
    gemm(xb, wb_1[l], M_TOK, 4096, 1024, EPI_GELU_BF16, nullptr, hbuf, L[l].bf1);
    gemm(hbuf, wb_2[l], M_TOK, 1024, 4096, EPI_F32, cbuf, nullptr, nullptr);
    ln_rows_kernel<<<M_TOK, TPB, 0, stream>>>(
        cbuf, L[l].bf2, xf, L[l].g2, L[l].b2, nullptr, nullptr, 2, xf, xb);
  }

  // --- pooling head ---
  build_attin_kernel<<<gs_blocks((size_t)4096 * 2048), TPB, 0, stream>>>(xf, attin);
  gemm(attin, wb_att, 4096, 1024, 2048, EPI_F32, cbuf, nullptr, nullptr);
  att_finish_kernel<<<4096, TPB, 0, stream>>>(
      cbuf, att1_b, gate_W, gate_b, aw_W, aw_b, sbuf);
  pool_kernel<<<BC, TPB, 0, stream>>>(xf, sbuf, cls_W, cls_b, (float*)d_out);
}
